// TabularResNet_67886253081200
// MI455X (gfx1250) — matrix operand order repaired and checked by reading, not compiled, not run on hardware
//
#include <hip/hip_runtime.h>

#define NROWS   16384
#define NNUM    32
#define NCAT    16
#define DM      256
#define DH      512
#define NLAYERS 8
#define KFIRST  12288   // 48 * 256

typedef __bf16 v16bf __attribute__((ext_vector_type(16)));
typedef __bf16 v8bf  __attribute__((ext_vector_type(8)));
typedef float  v8f   __attribute__((ext_vector_type(8)));
typedef unsigned short u16x8 __attribute__((ext_vector_type(8)));

__device__ __forceinline__ unsigned short f2bf(float f) {
    unsigned u = __float_as_uint(f);
    u += 0x7FFFu + ((u >> 16) & 1u);     // round-to-nearest-even
    return (unsigned short)(u >> 16);
}

// Pack 8 floats -> 8 bf16 and store as one 16-byte chunk.
__device__ __forceinline__ void store_bf8(unsigned short* dst, float4 a, float4 b) {
    u16x8 pk;
    pk[0] = f2bf(a.x); pk[1] = f2bf(a.y); pk[2] = f2bf(a.z); pk[3] = f2bf(a.w);
    pk[4] = f2bf(b.x); pk[5] = f2bf(b.y); pk[6] = f2bf(b.z); pk[7] = f2bf(b.w);
    *(u16x8*)dst = pk;
}

// A fragment, 16x32 bf16 tile from row-major buffer (row stride ld halves).
// Lane<16: row=lane, K {0..7} and {16..23}; lane>=16: row=lane-16, K {8..15} and {24..31}.
__device__ __forceinline__ v16bf load_frag_a(const unsigned short* base, int ld, int lane) {
    const unsigned short* p = base + (lane & 15) * ld + ((lane >> 4) << 3);
    v8bf lo = *(const v8bf*)p;
    v8bf hi = *(const v8bf*)(p + 16);
    return __builtin_shufflevector(lo, hi, 0,1,2,3,4,5,6,7,8,9,10,11,12,13,14,15);
}

// B fragment, 32x16 bf16 tile from K-major (N x K row-major) buffer.
// Lane<16: N=lane, K {0..7} and {16..23}; lane>=16: N=lane-16, K {8..15} and {24..31}.
__device__ __forceinline__ v16bf load_frag_b(const unsigned short* base, int ld, int lane) {
    const unsigned short* p = base + (lane & 15) * ld + ((lane >> 4) << 3);
    v8bf lo = *(const v8bf*)p;
    v8bf hi = *(const v8bf*)(p + 16);
    return __builtin_shufflevector(lo, hi, 0,1,2,3,4,5,6,7,8,9,10,11,12,13,14,15);
}

#define WMMA_BF16(a, b, c) \
    __builtin_amdgcn_wmma_f32_16x16x32_bf16(false, (a), false, (b), (short)0, (c), false, false)

// (cnt, K, N) f32 -> (cnt, N, K) bf16
__global__ void transpose_bf16_kernel(const float* __restrict__ in,
                                      unsigned short* __restrict__ out,
                                      int K, int N, int cnt) {
    size_t total = (size_t)cnt * K * N;
    for (size_t i = (size_t)blockIdx.x * blockDim.x + threadIdx.x; i < total;
         i += (size_t)gridDim.x * blockDim.x) {
        size_t mat = i / ((size_t)K * N);
        size_t rem = i - mat * (size_t)K * N;
        int n = (int)(rem / K);
        int k = (int)(rem - (size_t)n * K);
        out[i] = f2bf(in[mat * (size_t)K * N + (size_t)k * N + n]);
    }
}

// Fused embedding + first GEMM: h[64 rows x 256] per block, K = 12288.
// 8 waves, each owns a 32-column strip and iterates over all four 16-row tiles.
__global__ __launch_bounds__(256) void gemm_first_kernel(
    const float* __restrict__ xnum, const int* __restrict__ xcat,
    const float* __restrict__ w_num, const float* __restrict__ b_num,
    const float* __restrict__ cat_embed,
    const unsigned short* __restrict__ Wtf,   // (256, 12288) bf16 K-major
    const float* __restrict__ b_first,
    float* __restrict__ h) {
    __shared__ float          xv[64 * NNUM];
    __shared__ int            xid[64 * NCAT];
    __shared__ unsigned short Ast[64 * 128];

    const int t = threadIdx.x;
    const int lane = t & 31, w = t >> 5;
    const int row0 = blockIdx.x * 64;

    for (int i = t; i < 64 * NNUM; i += 256) xv[i] = xnum[(size_t)row0 * NNUM + i];
    for (int i = t; i < 64 * NCAT; i += 256) xid[i] = xcat[(size_t)row0 * NCAT + i];
    __syncthreads();

    v8f acc[4][2];
    {
        v8f z = {};
        for (int m = 0; m < 4; ++m) for (int n = 0; n < 2; ++n) acc[m][n] = z;
    }

    const int kk8  = (t & 15) * 8;   // 8 contiguous k positions per thread
    const int rb16 = t >> 4;         // 16 threads per row group

    for (int kc = 0; kc < KFIRST; kc += 128) {
        const int f = kc >> 8;               // feature index (128-chunk stays in one feature)
        const int d = (kc & 255) + kk8;      // offset within the 256-wide embedding dim
        if (f < NNUM) {
            const float4 w0 = *(const float4*)&w_num[f * DM + d];
            const float4 w1 = *(const float4*)&w_num[f * DM + d + 4];
            const float4 c0 = *(const float4*)&b_num[f * DM + d];
            const float4 c1 = *(const float4*)&b_num[f * DM + d + 4];
            #pragma unroll
            for (int i = 0; i < 4; ++i) {
                const int r = rb16 + 16 * i;
                const float x = xv[r * NNUM + f];
                float4 e0 = make_float4(x*w0.x+c0.x, x*w0.y+c0.y, x*w0.z+c0.z, x*w0.w+c0.w);
                float4 e1 = make_float4(x*w1.x+c1.x, x*w1.y+c1.y, x*w1.z+c1.z, x*w1.w+c1.w);
                store_bf8(&Ast[r * 128 + kk8], e0, e1);
            }
        } else {
            const int c = f - NNUM;
            #pragma unroll
            for (int i = 0; i < 4; ++i) {
                const int r = rb16 + 16 * i;
                const float* ep = cat_embed + (size_t)xid[r * NCAT + c] * DM + d;
                store_bf8(&Ast[r * 128 + kk8],
                          *(const float4*)ep, *(const float4*)(ep + 4));
            }
        }
        __syncthreads();

        #pragma unroll
        for (int k32 = 0; k32 < 128; k32 += 32) {
            v16bf b0 = load_frag_b(Wtf + (size_t)(w * 32     ) * KFIRST + kc + k32, KFIRST, lane);
            v16bf b1 = load_frag_b(Wtf + (size_t)(w * 32 + 16) * KFIRST + kc + k32, KFIRST, lane);
            #pragma unroll
            for (int m = 0; m < 4; ++m) {
                v16bf a = load_frag_a(&Ast[(m * 16) * 128 + k32], 128, lane);
                acc[m][0] = WMMA_BF16(a, b0, acc[m][0]);
                acc[m][1] = WMMA_BF16(a, b1, acc[m][1]);
            }
        }
        __syncthreads();
    }

    const int cn = lane & 15;
    const int roff = (lane >> 4) * 8;
    for (int n = 0; n < 2; ++n) {
        const int col = w * 32 + n * 16 + cn;
        const float bias = b_first[col];
        #pragma unroll
        for (int m = 0; m < 4; ++m) {
            #pragma unroll
            for (int i = 0; i < 8; ++i) {
                const int r = row0 + m * 16 + roff + i;
                h[(size_t)r * DM + col] = acc[m][n][i] + bias;
            }
        }
    }
}

// Fused residual block: LN -> GEMM(256->512)+ReLU -> GEMM(512->256) + residual.
// 32-row tile per block; each wave owns a column strip and walks both M-tiles.
__global__ __launch_bounds__(256) void resblock_kernel(
    float* __restrict__ h,
    const unsigned short* __restrict__ W1t,   // (512, 256) bf16 K-major
    const unsigned short* __restrict__ W2t,   // (256, 512) bf16 K-major
    const float* __restrict__ g, const float* __restrict__ bn,
    const float* __restrict__ b1, const float* __restrict__ b2) {
    __shared__ unsigned short aln[32 * DM];   // 16 KB
    __shared__ unsigned short hid[32 * DH];   // 32 KB

    const int t = threadIdx.x, lane = t & 31, w = t >> 5;
    const int row0 = blockIdx.x * 32;

    // LayerNorm: each wave handles 4 rows, 8 elems/lane, wave32 shuffle reduce.
    for (int rr = 0; rr < 4; ++rr) {
        const int r = w * 4 + rr;
        const float* hp = h + (size_t)(row0 + r) * DM;
        float v[8], s = 0.f, s2 = 0.f;
        #pragma unroll
        for (int i = 0; i < 8; ++i) { v[i] = hp[lane * 8 + i]; s += v[i]; s2 += v[i] * v[i]; }
        #pragma unroll
        for (int m = 16; m >= 1; m >>= 1) {
            s  += __shfl_xor(s,  m, 32);
            s2 += __shfl_xor(s2, m, 32);
        }
        const float mu  = s * (1.f / DM);
        const float var = s2 * (1.f / DM) - mu * mu;
        const float inv = rsqrtf(var + 1e-5f);
        #pragma unroll
        for (int i = 0; i < 8; ++i) {
            const int dd = lane * 8 + i;
            aln[r * DM + dd] = f2bf((v[i] - mu) * inv * g[dd] + bn[dd]);
        }
    }
    __syncthreads();

    // Stage 1: (32x256) @ (256x512) -> hid (bf16), bias + ReLU.
    // Wave w owns cols [w*64, w*64+64): 4 n-tiles x 2 m-tiles.
    {
        v8f acc[2][4];
        { v8f z = {}; for (int m = 0; m < 2; ++m) for (int n = 0; n < 4; ++n) acc[m][n] = z; }
        for (int ks = 0; ks < DM; ks += 32) {
            v16bf b[4];
            #pragma unroll
            for (int n = 0; n < 4; ++n)
                b[n] = load_frag_b(W1t + (size_t)(w * 64 + n * 16) * DM + ks, DM, lane);
            #pragma unroll
            for (int m = 0; m < 2; ++m) {
                v16bf a = load_frag_a(&aln[(m * 16) * DM + ks], DM, lane);
                #pragma unroll
                for (int n = 0; n < 4; ++n) acc[m][n] = WMMA_BF16(a, b[n], acc[m][n]);
            }
        }
        const int cn = lane & 15, roff = (lane >> 4) * 8;
        for (int n = 0; n < 4; ++n) {
            const int col = w * 64 + n * 16 + cn;
            const float bias = b1[col];
            #pragma unroll
            for (int m = 0; m < 2; ++m) {
                #pragma unroll
                for (int i = 0; i < 8; ++i) {
                    const float vv = acc[m][n][i] + bias;
                    hid[(m * 16 + roff + i) * DH + col] = f2bf(vv > 0.f ? vv : 0.f);
                }
            }
        }
    }
    __syncthreads();

    // Stage 2: (32x512) @ (512x256), bias + residual add, fp32 out in place.
    // Wave w owns cols [w*32, w*32+32): 2 n-tiles x 2 m-tiles.
    {
        v8f acc[2][2];
        { v8f z = {}; for (int m = 0; m < 2; ++m) for (int n = 0; n < 2; ++n) acc[m][n] = z; }
        for (int ks = 0; ks < DH; ks += 32) {
            v16bf b0 = load_frag_b(W2t + (size_t)(w * 32     ) * DH + ks, DH, lane);
            v16bf b1v = load_frag_b(W2t + (size_t)(w * 32 + 16) * DH + ks, DH, lane);
            #pragma unroll
            for (int m = 0; m < 2; ++m) {
                v16bf a = load_frag_a(&hid[(m * 16) * DH + ks], DH, lane);
                acc[m][0] = WMMA_BF16(a, b0,  acc[m][0]);
                acc[m][1] = WMMA_BF16(a, b1v, acc[m][1]);
            }
        }
        const int cn = lane & 15, roff = (lane >> 4) * 8;
        for (int n = 0; n < 2; ++n) {
            const int col = w * 32 + n * 16 + cn;
            const float bias = b2[col];
            #pragma unroll
            for (int m = 0; m < 2; ++m) {
                #pragma unroll
                for (int i = 0; i < 8; ++i) {
                    const int r = row0 + m * 16 + roff + i;
                    const size_t off = (size_t)r * DM + col;
                    h[off] = h[off] + acc[m][n][i] + bias;
                }
            }
        }
    }
}

// Final LN + head (D=256 -> 2): one row per wave.
__global__ __launch_bounds__(256) void head_kernel(
    const float* __restrict__ h, const float* __restrict__ gf,
    const float* __restrict__ bf, const float* __restrict__ Wh,
    const float* __restrict__ bh, float* __restrict__ out) {
    const int t = threadIdx.x, lane = t & 31, w = t >> 5;
    const int r = blockIdx.x * 8 + w;
    const float* hp = h + (size_t)r * DM;
    float v[8], s = 0.f, s2 = 0.f;
    #pragma unroll
    for (int i = 0; i < 8; ++i) { v[i] = hp[lane * 8 + i]; s += v[i]; s2 += v[i] * v[i]; }
    #pragma unroll
    for (int m = 16; m >= 1; m >>= 1) {
        s  += __shfl_xor(s,  m, 32);
        s2 += __shfl_xor(s2, m, 32);
    }
    const float mu  = s * (1.f / DM);
    const float var = s2 * (1.f / DM) - mu * mu;
    const float inv = rsqrtf(var + 1e-5f);
    float p0 = 0.f, p1 = 0.f;
    #pragma unroll
    for (int i = 0; i < 8; ++i) {
        const int d = lane * 8 + i;
        const float x = (v[i] - mu) * inv * gf[d] + bf[d];
        p0 += x * Wh[d * 2 + 0];
        p1 += x * Wh[d * 2 + 1];
    }
    #pragma unroll
    for (int m = 16; m >= 1; m >>= 1) {
        p0 += __shfl_xor(p0, m, 32);
        p1 += __shfl_xor(p1, m, 32);
    }
    if (lane == 0) {
        out[(size_t)r * 2 + 0] = p0 + bh[0];
        out[(size_t)r * 2 + 1] = p1 + bh[1];
    }
}

extern "C" void kernel_launch(void* const* d_in, const int* in_sizes, int n_in,
                              void* d_out, int out_size, void* d_ws, size_t ws_size,
                              hipStream_t stream) {
    const float* xnum      = (const float*)d_in[0];
    const int*   xcat      = (const int*)  d_in[1];
    const float* w_num     = (const float*)d_in[2];
    const float* b_num     = (const float*)d_in[3];
    const float* cat_embed = (const float*)d_in[4];
    const float* W_first   = (const float*)d_in[5];
    const float* b_first   = (const float*)d_in[6];
    const float* ln_g      = (const float*)d_in[7];
    const float* ln_b      = (const float*)d_in[8];
    const float* W1s       = (const float*)d_in[9];
    const float* b1s       = (const float*)d_in[10];
    const float* W2s       = (const float*)d_in[11];
    const float* b2s       = (const float*)d_in[12];
    const float* g_f       = (const float*)d_in[13];
    const float* beta_f    = (const float*)d_in[14];
    const float* W_head    = (const float*)d_in[15];
    const float* b_head    = (const float*)d_in[16];
    float* out = (float*)d_out;

    // Workspace layout (~27.3 MB)
    unsigned short* Wtf = (unsigned short*)d_ws;                 // 256*12288 bf16
    unsigned short* W1t = Wtf + (size_t)DM * KFIRST;             // 8*512*256 bf16
    unsigned short* W2t = W1t + (size_t)NLAYERS * DH * DM;       // 8*256*512 bf16
    float* h = (float*)(W2t + (size_t)NLAYERS * DM * DH);        // 16384*256 f32

    transpose_bf16_kernel<<<1024, 256, 0, stream>>>(W_first, Wtf, KFIRST, DM, 1);
    transpose_bf16_kernel<<<512, 256, 0, stream>>>(W1s, W1t, DM, DH, NLAYERS);
    transpose_bf16_kernel<<<512, 256, 0, stream>>>(W2s, W2t, DH, DM, NLAYERS);

    gemm_first_kernel<<<NROWS / 64, 256, 0, stream>>>(
        xnum, xcat, w_num, b_num, cat_embed, Wtf, b_first, h);

    for (int l = 0; l < NLAYERS; ++l) {
        resblock_kernel<<<NROWS / 32, 256, 0, stream>>>(
            h, W1t + (size_t)l * DH * DM, W2t + (size_t)l * DM * DH,
            ln_g + l * DM, ln_b + l * DM, b1s + l * DH, b2s + l * DM);
    }

    head_kernel<<<NROWS / 8, 256, 0, stream>>>(h, g_f, beta_f, W_head, b_head, out);
}
